// HeteroEncoder_30305289240582
// MI455X (gfx1250) — matrix operand order repaired and checked by reading, not compiled, not run on hardware
//
#include <hip/hip_runtime.h>
#include <stdint.h>

typedef _Float16 half_t;
typedef __attribute__((ext_vector_type(16))) _Float16 v16h;
typedef __attribute__((ext_vector_type(8)))  _Float16 v8h;
typedef __attribute__((ext_vector_type(8)))  float    v8f;
typedef __attribute__((ext_vector_type(4)))  float    v4f;

#define CST 260            // C (f32) LDS row stride: 256+4 to dodge bank conflicts
#define LN_EPS 1e-5f

// ---------------------------------------------------------------------------
// wave32 full-lane reduction
__device__ __forceinline__ float wave_sum(float v) {
#pragma unroll
  for (int off = 16; off > 0; off >>= 1) v += __shfl_xor(v, off, 32);
  return v;
}

// ---------------------------------------------------------------------------
// 64xK (LDS, f16) @ KxN=256 (packed f16 global) -> 64x256 f32 into LDS Cbuf.
// Block = 256 threads = 8 waves. Wave w owns n-tiles {2w, 2w+1} across all 4
// row strips -> every B fragment is read exactly once per block. The two B
// fragments for ko+1 are double-buffered in registers; KCHUNKS is a template
// constant so the ko loop fully unrolls and register renaming (not v_movs)
// implements the double buffer, letting the scheduler hoist B loads into
// clauses well ahead of their consuming WMMAs (round-2-verified codegen).
// Packed B layout: fragment (ko,nt) at Bp + ((ko*16+nt)*512 + lane*16),
// lane -> (col = nt*16 + lane%16, h = lane>=16), half j -> K = 8h + j%8 + (j/8)*16.
template <int KCHUNKS>
__device__ __forceinline__ void gemm_tile(const half_t* __restrict__ As, int astride,
                                          const half_t* __restrict__ Bp,
                                          float* __restrict__ Cb) {
  const int tid = threadIdx.x, wave = tid >> 5, lane = tid & 31;
  const int nt0 = wave << 1;            // 2 n-tiles per wave
  v8f acc[4][2] = {};
  const int arow_l = lane & 15;
  const int abase  = (lane >> 4) << 3;  // lanes 16-31 start at K+8 per ISA A layout
  const half_t* bptr = Bp + ((size_t)nt0 << 9) + (lane << 4);
  v16h b0 = *(const v16h*)(bptr);
  v16h b1 = *(const v16h*)(bptr + 512);
#pragma unroll
  for (int ko = 0; ko < KCHUNKS; ++ko) {
    v16h nb0 = b0, nb1 = b1;
    if (ko + 1 < KCHUNKS) {
      const half_t* np = bptr + (size_t)(ko + 1) * 8192;
      nb0 = *(const v16h*)(np);
      nb1 = *(const v16h*)(np + 512);
    }
#pragma unroll
    for (int rs = 0; rs < 4; ++rs) {
      const half_t* ap = As + (rs * 16 + arow_l) * astride + (ko << 5) + abase;
      v8h lo = *(const v8h*)(ap);
      v8h hi = *(const v8h*)(ap + 16);
      v16h a;
#pragma unroll
      for (int i = 0; i < 8; ++i) { a[i] = lo[i]; a[i + 8] = hi[i]; }
      acc[rs][0] = __builtin_amdgcn_wmma_f32_16x16x32_f16(false, a, false, b0,
                                                          (short)0, acc[rs][0], false, false);
      acc[rs][1] = __builtin_amdgcn_wmma_f32_16x16x32_f16(false, a, false, b1,
                                                          (short)0, acc[rs][1], false, false);
    }
    b0 = nb0; b1 = nb1;
  }
  // C layout: VGPR r, lanes 0-15 -> M=r, lanes 16-31 -> M=r+8; col = lane%16
  const int crow_l = (lane >> 4) << 3;
  const int ccol   = lane & 31 & 15;
#pragma unroll
  for (int rs = 0; rs < 4; ++rs) {
#pragma unroll
    for (int n = 0; n < 2; ++n) {
      float* cp = Cb + (rs * 16 + crow_l) * CST + ((nt0 + n) << 4) + ccol;
#pragma unroll
      for (int r = 0; r < 8; ++r) cp[r * CST] = acc[rs][n][r];
    }
  }
}

// ---------------------------------------------------------------------------
// LayerNorm(x + bias)*g + be, then ReLU (TANH=0) or tanh (TANH=1).
// Wave-per-row: 8 waves x 8 rows each; lane holds cols {j*32+lane}.
template <int TANH>
__device__ __forceinline__ void ln_act_to_lds(const float* Cb, const float* bias,
                                              const float* g, const float* be,
                                              half_t* Aout, int astride) {
  const int wave = threadIdx.x >> 5, lane = threadIdx.x & 31;
  float bv[8], gv[8], bev[8];
#pragma unroll
  for (int j = 0; j < 8; ++j) {
    int c = (j << 5) + lane;
    bv[j] = bias[c]; gv[j] = g[c]; bev[j] = be[c];
  }
  for (int r = wave; r < 64; r += 8) {
    const float* row = Cb + r * CST;
    float v[8]; float s = 0.f;
#pragma unroll
    for (int j = 0; j < 8; ++j) { v[j] = row[(j << 5) + lane] + bv[j]; s += v[j]; }
    s = wave_sum(s);
    const float mu = s * (1.f / 256.f);
    float q = 0.f;
#pragma unroll
    for (int j = 0; j < 8; ++j) { float d = v[j] - mu; q += d * d; }
    q = wave_sum(q);
    const float rstd = rsqrtf(q * (1.f / 256.f) + LN_EPS);
#pragma unroll
    for (int j = 0; j < 8; ++j) {
      float y = (v[j] - mu) * rstd * gv[j] + bev[j];
      y = TANH ? tanhf(y) : fmaxf(y, 0.f);
      Aout[r * astride + (j << 5) + lane] = (half_t)y;
    }
  }
}

// final layer for nodes: rows are bin-list entries (nid), wr flags row validity
__device__ __forceinline__ void ln_tanh_node_out(const float* Cb, const float* bias,
                                                 const float* g, const float* be,
                                                 float* outn, half_t* cache,
                                                 const int* nid, const int* wr) {
  const int wave = threadIdx.x >> 5, lane = threadIdx.x & 31;
  float bv[8], gv[8], bev[8];
#pragma unroll
  for (int j = 0; j < 8; ++j) {
    int c = (j << 5) + lane;
    bv[j] = bias[c]; gv[j] = g[c]; bev[j] = be[c];
  }
  for (int r = wave; r < 64; r += 8) {
    if (!wr[r]) continue;                       // uniform within the wave
    const int node = nid[r];
    const float* row = Cb + r * CST;
    float v[8]; float s = 0.f;
#pragma unroll
    for (int j = 0; j < 8; ++j) { v[j] = row[(j << 5) + lane] + bv[j]; s += v[j]; }
    s = wave_sum(s);
    const float mu = s * (1.f / 256.f);
    float q = 0.f;
#pragma unroll
    for (int j = 0; j < 8; ++j) { float d = v[j] - mu; q += d * d; }
    q = wave_sum(q);
    const float rstd = rsqrtf(q * (1.f / 256.f) + LN_EPS);
#pragma unroll
    for (int j = 0; j < 8; ++j) {
      int c = (j << 5) + lane;
      float y = tanhf((v[j] - mu) * rstd * gv[j] + bev[j]);
      if (cache) {
        // f32 copy is never re-read on device: stream past the caches
        __builtin_nontemporal_store(y, &outn[(size_t)node * 256 + c]);
        cache[(size_t)node * 256 + c] = (half_t)y;   // re-read by gather: keep RT
      } else {
        outn[(size_t)node * 256 + c] = y;            // gather reads this: keep RT
      }
    }
  }
}

__device__ __forceinline__ void ln_tanh_edge_out(const float* Cb, const float* bias,
                                                 const float* g, const float* be,
                                                 float* oute, const int* eids) {
  const int wave = threadIdx.x >> 5, lane = threadIdx.x & 31;
  float bv[8], gv[8], bev[8];
#pragma unroll
  for (int j = 0; j < 8; ++j) {
    int c = (j << 5) + lane;
    bv[j] = bias[c]; gv[j] = g[c]; bev[j] = be[c];
  }
  for (int r = wave; r < 64; r += 8) {
    const int e = eids[r];
    if (e < 0) continue;
    const float* row = Cb + r * CST;
    float v[8]; float s = 0.f;
#pragma unroll
    for (int j = 0; j < 8; ++j) { v[j] = row[(j << 5) + lane] + bv[j]; s += v[j]; }
    s = wave_sum(s);
    const float mu = s * (1.f / 256.f);
    float q = 0.f;
#pragma unroll
    for (int j = 0; j < 8; ++j) { float d = v[j] - mu; q += d * d; }
    q = wave_sum(q);
    const float rstd = rsqrtf(q * (1.f / 256.f) + LN_EPS);
#pragma unroll
    for (int j = 0; j < 8; ++j) {
      int c = (j << 5) + lane;
      float y = tanhf((v[j] - mu) * rstd * gv[j] + bev[j]);
      __builtin_nontemporal_store(y, &oute[(size_t)e * 256 + c]);  // 512MB stream
    }
  }
}

// ---------------------------------------------------------------------------
// f32 [K,256] -> f16 WMMA-B packed layout
__global__ __launch_bounds__(256) void pack_w_kernel(const float* __restrict__ W,
                                                     half_t* __restrict__ dst, int K) {
  int p = blockIdx.x * 256 + threadIdx.x;
  if (p >= K * 256) return;
  int j    = p & 15;
  int lane = (p >> 4) & 31;
  int nt   = (p >> 9) & 15;
  int ko   = p >> 13;
  int col  = nt * 16 + (lane & 15);
  int kk   = ko * 32 + ((lane >> 4) << 3) + (j & 7) + ((j >> 3) << 4);
  dst[p] = (half_t)W[(size_t)kk * 256 + col];
}

// ---------------------------------------------------------------------------
// bin nodes by model, edges by combo
__global__ __launch_bounds__(256) void classify_nodes_kernel(const int* __restrict__ vid,
                                                             int* counters, int* nlists, int N) {
  int n = blockIdx.x * 256 + threadIdx.x;
  if (n >= N) return;
  int m = vid[n] >= 3;
  int pos = atomicAdd(&counters[4 + m], 1);
  nlists[(size_t)m * N + pos] = n;
}

__global__ __launch_bounds__(256) void classify_edges_kernel(const int* __restrict__ ei,
                                                             const int* __restrict__ vid,
                                                             int* counters, int* lists, int E) {
  int e = blockIdx.x * 256 + threadIdx.x;
  if (e >= E) return;
  int a = vid[ei[e]] >= 3;
  int b = vid[ei[E + e]] >= 3;
  int idx = (a << 1) | b;
  int combo = (idx == 0) ? 0 : (idx == 1) ? 1 : (idx == 3) ? 2 : 3; // 3 = unmatched
  int pos = atomicAdd(&counters[combo], 1);
  lists[(size_t)combo * E + pos] = e;
}

__global__ __launch_bounds__(256) void zero_unmatched_kernel(const int* counters,
                                                             const int* lists,
                                                             float* oute, int E) {
  int cnt = counters[3];
  int base = blockIdx.x * 8;
  if (base >= cnt) return;
  for (int i = 0; i < 8; ++i) {
    int idx = base + i;
    if (idx >= cnt) break;
    int e = lists[(size_t)3 * E + idx];
    __builtin_nontemporal_store(0.f, &oute[(size_t)e * 256 + threadIdx.x]);
  }
}

// ---------------------------------------------------------------------------
struct NodeArgs {
  const float* x;
  const int* counts;   // [4+m]
  const int* nlists;   // 2*N
  const float* W0[2];
  const float* b[2][3]; const float* g[2][3]; const float* be[2][3];
  const half_t* pW1[2]; const half_t* pW2[2];
  float* outn; half_t* cache;
  int N; int xstride; int nf[2];
};

__global__ __launch_bounds__(256) void node_mlp_kernel(NodeArgs A) {
  extern __shared__ char smem[];
  int*    nid = (int*)smem;                               // 64 (clamped node ids)
  int*    wr  = (int*)(smem + 256);                       // 64 (validity)
  half_t* Ab  = (half_t*)(smem + 512);                    // 64 x 264 f16
  float*  Cb  = (float*)(smem + 512 + 64 * 264 * 2);      // 64 x 260 f32
  const int tid = threadIdx.x;
  const int m = blockIdx.y;
  const int cnt = A.counts[4 + m];
  const int tile0 = blockIdx.x * 64;
  if (tile0 >= cnt) return;

  if (tid < 64) {
    int idx = tile0 + tid;
    int cl = idx < cnt ? idx : cnt - 1;
    nid[tid] = A.nlists[(size_t)m * A.N + cl];
    wr[tid]  = idx < cnt;
  }
  __syncthreads();

  // layer 0: tiny K (3 or 6), scalar
  const int nf = A.nf[m];
  const float* W0 = A.W0[m];
  for (int it = tid; it < 64 * 256; it += 256) {
    int r = it >> 8, c = it & 255;
    int node = nid[r];
    float acc = 0.f;
    for (int f = 0; f < nf; ++f) acc += A.x[(size_t)node * A.xstride + f] * W0[f * 256 + c];
    Cb[r * CST + c] = acc;
  }
  __syncthreads();
  ln_act_to_lds<0>(Cb, A.b[m][0], A.g[m][0], A.be[m][0], Ab, 264);
  __syncthreads();
  gemm_tile<8>(Ab, 264, A.pW1[m], Cb);
  __syncthreads();
  ln_act_to_lds<0>(Cb, A.b[m][1], A.g[m][1], A.be[m][1], Ab, 264);
  __syncthreads();
  gemm_tile<8>(Ab, 264, A.pW2[m], Cb);
  __syncthreads();
  ln_tanh_node_out(Cb, A.b[m][2], A.g[m][2], A.be[m][2], A.outn, A.cache, nid, wr);
}

// ---------------------------------------------------------------------------
struct EdgeArgs {
  const int* ei; const int* counts; const int* lists;
  const float* b[3][3]; const float* g[3][3]; const float* be[3][3];
  const half_t* pW0[3]; const half_t* pW1[3]; const half_t* pW2[3];
  const float* nodes_f32; const half_t* cache;
  float* oute;
  int E; int use16;
};

__global__ __launch_bounds__(256) void edge_mlp_kernel(EdgeArgs A) {
  extern __shared__ char smem[];
  int*    eids = (int*)smem;                 // 64
  int*    ns   = (int*)(smem + 256);         // 64
  int*    nd   = (int*)(smem + 512);         // 64
  half_t* Ab   = (half_t*)(smem + 1024);     // 64 x 520 f16 (K=512 + pad 8)
  float*  Cb   = (float*)(smem + 1024 + 64 * 520 * 2); // 64 x 260 f32
  const int tid = threadIdx.x;
  const int k = blockIdx.y;
  const int tile0 = blockIdx.x * 64;
  const int cnt = A.counts[k];
  if (tile0 >= cnt) return;

  if (tid < 64) {
    int idx = tile0 + tid;
    int cl = idx < cnt ? idx : cnt - 1;
    int e = A.lists[(size_t)k * A.E + cl];
    eids[tid] = (idx < cnt) ? e : -1;
    ns[tid] = A.ei[e];
    nd[tid] = A.ei[A.E + e];
  }
  __syncthreads();

  // gather 64 rows x 512 halves; one wave covers 512 contiguous bytes per node
  // row. f16 path: CDNA5 async global->LDS copy (ASYNCcnt), no VGPR roundtrip.
  if (A.use16) {
    for (int it = tid; it < 64 * 64; it += 256) {   // uniform trip count: EXEC all 1s
      int r = it >> 6, c = it & 63;
      int node = (c < 32) ? ns[r] : nd[r];
      int cc = (c & 31) << 3;
      half_t* dp = Ab + r * 520 + (c << 3);
      const half_t* sp = A.cache + (size_t)node * 256 + cc;
      uint32_t lds_addr = (uint32_t)(uintptr_t)dp;
      asm volatile("global_load_async_to_lds_b128 %0, %1, off"
                   :: "v"(lds_addr), "v"(sp) : "memory");
    }
    asm volatile("s_wait_asynccnt 0" ::: "memory");
  } else {
    for (int it = tid; it < 64 * 64; it += 256) {
      int r = it >> 6, c = it & 63;
      int node = (c < 32) ? ns[r] : nd[r];
      int cc = (c & 31) << 3;
      half_t* dp = Ab + r * 520 + (c << 3);
      const float* sp = A.nodes_f32 + (size_t)node * 256 + cc;
      v4f f0 = *(const v4f*)sp, f1 = *(const v4f*)(sp + 4);
      v8h h;
#pragma unroll
      for (int i = 0; i < 4; ++i) { h[i] = (half_t)f0[i]; h[i + 4] = (half_t)f1[i]; }
      *(v8h*)dp = h;
    }
  }
  __syncthreads();
  gemm_tile<16>(Ab, 520, A.pW0[k], Cb);
  __syncthreads();
  ln_act_to_lds<0>(Cb, A.b[k][0], A.g[k][0], A.be[k][0], Ab, 264);
  __syncthreads();
  gemm_tile<8>(Ab, 264, A.pW1[k], Cb);
  __syncthreads();
  ln_act_to_lds<0>(Cb, A.b[k][1], A.g[k][1], A.be[k][1], Ab, 264);
  __syncthreads();
  gemm_tile<8>(Ab, 264, A.pW2[k], Cb);
  __syncthreads();
  ln_tanh_edge_out(Cb, A.b[k][2], A.g[k][2], A.be[k][2], A.oute, eids);
}

// ---------------------------------------------------------------------------
extern "C" void kernel_launch(void* const* d_in, const int* in_sizes, int n_in,
                              void* d_out, int out_size, void* d_ws, size_t ws_size,
                              hipStream_t stream) {
  const float* x   = (const float*)d_in[0];
  const int*   ei  = (const int*)d_in[1];
  const int*   vid = (const int*)d_in[2];
  const int N = in_sizes[2];
  const int E = in_sizes[1] / 2;
  const int xstride = in_sizes[0] / N;

  // d_in layout: 0:x 1:edge_index 2:volume_id, then node_params (2x3x4), edge_params (3x3x4)
  uint8_t* w = (uint8_t*)d_ws;
  int* counters = (int*)w;                       // [0..3] edge combos, [4..5] node models
  int* elists   = (int*)(w + 256);               // 4 * E ints
  size_t cur = 256 + (size_t)4 * E * sizeof(int);
  int* nlists   = (int*)(w + cur);               // 2 * N ints
  cur += (size_t)2 * N * sizeof(int);
  cur = (cur + 255) & ~(size_t)255;
  auto alloc_h = [&](size_t halves) -> half_t* {
    half_t* p = (half_t*)(w + cur);
    cur += halves * 2;
    cur = (cur + 255) & ~(size_t)255;
    return p;
  };
  half_t *npW1[2], *npW2[2];
  for (int m = 0; m < 2; ++m) { npW1[m] = alloc_h(65536); npW2[m] = alloc_h(65536); }
  half_t *epW0[3], *epW1[3], *epW2[3];
  for (int k = 0; k < 3; ++k) {
    epW0[k] = alloc_h(131072); epW1[k] = alloc_h(65536); epW2[k] = alloc_h(65536);
  }
  half_t* cache = nullptr;
  size_t cache_bytes = (size_t)N * 256 * 2;
  if (cur + cache_bytes <= ws_size) { cache = (half_t*)(w + cur); cur += cache_bytes; }

  float* outn = (float*)d_out;
  float* oute = outn + (size_t)N * 256;

  hipMemsetAsync(counters, 0, 256, stream);

  // pack f32 weights into f16 WMMA-B fragments (only K in {256,512})
  for (int m = 0; m < 2; ++m) {
    pack_w_kernel<<<256, 256, 0, stream>>>((const float*)d_in[3 + m * 12 + 4], npW1[m], 256);
    pack_w_kernel<<<256, 256, 0, stream>>>((const float*)d_in[3 + m * 12 + 8], npW2[m], 256);
  }
  for (int k = 0; k < 3; ++k) {
    pack_w_kernel<<<512, 256, 0, stream>>>((const float*)d_in[27 + k * 12 + 0], epW0[k], 512);
    pack_w_kernel<<<256, 256, 0, stream>>>((const float*)d_in[27 + k * 12 + 4], epW1[k], 256);
    pack_w_kernel<<<256, 256, 0, stream>>>((const float*)d_in[27 + k * 12 + 8], epW2[k], 256);
  }

  classify_nodes_kernel<<<(N + 255) / 256, 256, 0, stream>>>(vid, counters, nlists, N);
  classify_edges_kernel<<<(E + 255) / 256, 256, 0, stream>>>(ei, vid, counters, elists, E);

  NodeArgs na{};
  na.x = x; na.counts = counters; na.nlists = nlists;
  na.outn = outn; na.cache = cache;
  na.N = N; na.xstride = xstride;
  for (int m = 0; m < 2; ++m) {
    na.W0[m] = (const float*)d_in[3 + m * 12 + 0];
    na.nf[m] = in_sizes[3 + m * 12 + 0] / 256;
    for (int l = 0; l < 3; ++l) {
      na.b[m][l]  = (const float*)d_in[3 + m * 12 + l * 4 + 1];
      na.g[m][l]  = (const float*)d_in[3 + m * 12 + l * 4 + 2];
      na.be[m][l] = (const float*)d_in[3 + m * 12 + l * 4 + 3];
    }
    na.pW1[m] = npW1[m]; na.pW2[m] = npW2[m];
  }
  const int node_smem = 512 + 64 * 264 * 2 + 64 * CST * 4;
  node_mlp_kernel<<<dim3((N + 63) / 64, 2), 256, node_smem, stream>>>(na);

  zero_unmatched_kernel<<<(E + 7) / 8, 256, 0, stream>>>(counters, elists, oute, E);

  EdgeArgs ea{};
  ea.ei = ei; ea.counts = counters; ea.lists = elists;
  ea.nodes_f32 = outn; ea.cache = cache; ea.oute = oute;
  ea.E = E; ea.use16 = (cache != nullptr) ? 1 : 0;
  for (int k = 0; k < 3; ++k) {
    for (int l = 0; l < 3; ++l) {
      ea.b[k][l]  = (const float*)d_in[27 + k * 12 + l * 4 + 1];
      ea.g[k][l]  = (const float*)d_in[27 + k * 12 + l * 4 + 2];
      ea.be[k][l] = (const float*)d_in[27 + k * 12 + l * 4 + 3];
    }
    ea.pW0[k] = epW0[k]; ea.pW1[k] = epW1[k]; ea.pW2[k] = epW2[k];
  }
  const int edge_smem = 1024 + 64 * 520 * 2 + 64 * CST * 4;
  edge_mlp_kernel<<<dim3((E + 63) / 64, 3), 256, edge_smem, stream>>>(ea);
}
